// MoEMinGRULayer_80530636800509
// MI455X (gfx1250) — matrix operand order repaired and checked by reading, not compiled, not run on hardware
//
#include <hip/hip_runtime.h>
#include <cstdint>

#define DMODEL 1024
#define SEQ    4096
#define BATCH  4
#define NEXP   4
#define XSPITCH (DMODEL + 8)   // LDS pitch in bf16 elements (bank-conflict pad)

typedef __attribute__((ext_vector_type(16))) __bf16 bf16x16;
typedef __attribute__((ext_vector_type(8)))  __bf16 bf16x8;
typedef __attribute__((ext_vector_type(8)))  float  f32x8;

__device__ __forceinline__ uint16_t bfbits(float f) {
  uint32_t u = __builtin_bit_cast(uint32_t, f);
  u += 0x7FFFu + ((u >> 16) & 1u);                 // round-to-nearest-even
  return (uint16_t)(u >> 16);
}
__device__ __forceinline__ float sigf(float v) { return 1.0f / (1.0f + __expf(-v)); }
__device__ __forceinline__ float tanhfast(float v) {
  v = fminf(fmaxf(v, -15.0f), 15.0f);
  float e2 = __expf(2.0f * v);
  return (e2 - 1.0f) / (e2 + 1.0f);
}

// ---------------------------------------------------------------------------
// One-shot fp32 -> bf16 conversion (weights and x), 4 elements / thread.
// Removes all conversion VALU from the hot WMMA loop.
// ---------------------------------------------------------------------------
__global__ void __launch_bounds__(256)
cvt_bf16_kernel(const float* __restrict__ src, uint16_t* __restrict__ dst, int n4) {
  const int i = blockIdx.x * blockDim.x + threadIdx.x;
  if (i >= n4) return;
  const float4 f = ((const float4*)src)[i];
  uint2 p;
  p.x = (uint32_t)bfbits(f.x) | ((uint32_t)bfbits(f.y) << 16);
  p.y = (uint32_t)bfbits(f.z) | ((uint32_t)bfbits(f.w) << 16);
  ((uint2*)dst)[i] = p;
}

// ---------------------------------------------------------------------------
// Router: gate logits (GEMV, E=4), top-2, softmax over selected.
// Writes dense per-token weights gw[token][e] (0 for unselected experts).
// ---------------------------------------------------------------------------
__global__ void __launch_bounds__(256)
router_topk_kernel(const float* __restrict__ x,
                   const float* __restrict__ Wgate,
                   float* __restrict__ gw) {
  const int wid   = threadIdx.x >> 5;
  const int lane  = threadIdx.x & 31;
  const int token = blockIdx.x * (blockDim.x >> 5) + wid;
  if (token >= BATCH * SEQ) return;

  const float* xp = x + (size_t)token * DMODEL;
  float a0 = 0.f, a1 = 0.f, a2 = 0.f, a3 = 0.f;
  for (int i = lane; i < DMODEL; i += 32) {
    const float xv = xp[i];
    a0 = fmaf(xv, Wgate[0 * DMODEL + i], a0);
    a1 = fmaf(xv, Wgate[1 * DMODEL + i], a1);
    a2 = fmaf(xv, Wgate[2 * DMODEL + i], a2);
    a3 = fmaf(xv, Wgate[3 * DMODEL + i], a3);
  }
#pragma unroll
  for (int off = 16; off > 0; off >>= 1) {
    a0 += __shfl_xor(a0, off, 32);
    a1 += __shfl_xor(a1, off, 32);
    a2 += __shfl_xor(a2, off, 32);
    a3 += __shfl_xor(a3, off, 32);
  }
  if (lane == 0) {
    float l[NEXP] = {a0, a1, a2, a3};
    int i0 = 0;
#pragma unroll
    for (int e = 1; e < NEXP; ++e) if (l[e] > l[i0]) i0 = e;
    int i1 = -1;
#pragma unroll
    for (int e = 0; e < NEXP; ++e)
      if (e != i0 && (i1 < 0 || l[e] > l[i1])) i1 = e;
    const float m  = fmaxf(l[i0], l[i1]);
    const float e0 = __expf(l[i0] - m);
    const float e1 = __expf(l[i1] - m);
    const float inv = 1.0f / (e0 + e1);
    float wout[NEXP] = {0.f, 0.f, 0.f, 0.f};
    wout[i0] = e0 * inv;
    wout[i1] = e1 * inv;
    float* g = gw + (size_t)token * NEXP;
#pragma unroll
    for (int e = 0; e < NEXP; ++e) g[e] = wout[e];
  }
}

// ---------------------------------------------------------------------------
// Fused per-expert kernel (all operands pre-converted bf16):
//   WMMA projections (g,v,d) + activations + causal scan + weighted scatter.
// grid  = NEXP * BATCH * (DMODEL/64) = 256 blocks, 128 threads (4 wave32).
// x chunk staged into LDS via CDNA5 async global->LDS copies (ASYNCcnt).
// ---------------------------------------------------------------------------
__global__ void __launch_bounds__(128)
moe_mingru_expert_kernel(const __bf16* __restrict__ xbf,
                         const __bf16* __restrict__ wgp0,
                         const __bf16* __restrict__ wvp0,
                         const __bf16* __restrict__ wdp0,
                         const float*  __restrict__ bg,
                         const float*  __restrict__ bv,
                         const float*  __restrict__ bd,
                         const float*  __restrict__ gw,
                         float* __restrict__ out) {
  __shared__ __align__(16) __bf16 xs[16 * XSPITCH];   // ~33 KB staged x chunk
  __shared__ float As[4][16][17];                     // decay a_t  (per-wave tile)
  __shared__ float Xs[4][16][17];                     // input x_t  (per-wave tile)

  const int cb   = blockIdx.x & 15;          // 64-column block
  const int b    = (blockIdx.x >> 4) & 3;    // batch
  const int e    = blockIdx.x >> 6;          // expert
  const int w    = threadIdx.x >> 5;         // wave in block
  const int lane = threadIdx.x & 31;
  const int half = lane >> 4;
  const int l16  = lane & 15;
  const int colbase = cb * 64 + w * 16;
  const int col     = colbase + l16;

  const __bf16* wgp = wgp0 + (size_t)e * DMODEL * DMODEL;
  const __bf16* wvp = wvp0 + (size_t)e * DMODEL * DMODEL;
  const __bf16* wdp = wdp0 + (size_t)e * DMODEL * DMODEL;
  const float   bgc = bg[(size_t)e * DMODEL + col];
  const float   bvc = bv[(size_t)e * DMODEL + col];
  const float   bdc = bd[(size_t)e * DMODEL + col];

  const uint32_t lbase = (uint32_t)(uintptr_t)(void*)xs;  // LDS byte offset

  float hcarry = 0.0f;   // scan carry, lanes 0..15 (one output column each)

  for (int s0 = 0; s0 < SEQ; s0 += 16) {
    __syncthreads();   // previous chunk's LDS consumers are done

    // --- async-copy 16x1024 bf16 x rows into LDS (no VGPR round trip) -----
    {
      const __bf16* xg = xbf + ((size_t)b * SEQ + s0) * DMODEL;
      for (int i = threadIdx.x; i < 16 * DMODEL / 8; i += blockDim.x) {
        const int row = i >> 7;                 // 128 x 16B segments per row
        const int seg = i & 127;
        const uint32_t laddr =
            lbase + (uint32_t)((row * XSPITCH + seg * 8) * 2);
        const uint64_t gaddr =
            (uint64_t)(uintptr_t)(xg + row * DMODEL + seg * 8);
        asm volatile("global_load_async_to_lds_b128 %0, %1, off"
                     :: "v"(laddr), "v"(gaddr) : "memory");
      }
      asm volatile("s_wait_asynccnt 0x0" ::: "memory");
    }
    __syncthreads();

    // --- accumulators initialized with bias -------------------------------
    f32x8 accg, accv, accd;
#pragma unroll
    for (int j = 0; j < 8; ++j) { accg[j] = bgc; accv[j] = bvc; accd[j] = bdc; }

    const int rowA = l16;   // A-matrix row this lane holds (16x32 bf16 layout)
    for (int kb = 0; kb < DMODEL / 32; ++kb) {
      // A tile from LDS: K = kb*32 + half*16 + 0..15
      const int ka = kb * 32 + half * 16;
      union { bf16x16 v; bf16x8 h[2]; } af;
      af.h[0] = *(const bf16x8*)&xs[rowA * XSPITCH + ka];
      af.h[1] = *(const bf16x8*)&xs[rowA * XSPITCH + ka + 8];

      // B tiles (32x16 bf16): lane -> col, K = kb*32 + half*16 + 0..15
      const size_t wofs = (size_t)col * DMODEL + kb * 32 + half * 16;
      if (kb + 1 < DMODEL / 32) {
        __builtin_prefetch(wgp + wofs + 32, 0, 1);
        __builtin_prefetch(wvp + wofs + 32, 0, 1);
        __builtin_prefetch(wdp + wofs + 32, 0, 1);
      }
      const bf16x16 bmg = *(const bf16x16*)(wgp + wofs);
      const bf16x16 bmv = *(const bf16x16*)(wvp + wofs);
      const bf16x16 bmd = *(const bf16x16*)(wdp + wofs);

      accg = __builtin_amdgcn_wmma_f32_16x16x32_bf16(
          false, af.v, false, bmg, (short)0, accg, false, false);
      accv = __builtin_amdgcn_wmma_f32_16x16x32_bf16(
          false, af.v, false, bmv, (short)0, accv, false, false);
      accd = __builtin_amdgcn_wmma_f32_16x16x32_bf16(
          false, af.v, false, bmd, (short)0, accd, false, false);
    }

    // --- activations -> per-wave (time x col) tiles in LDS ----------------
#pragma unroll
    for (int j = 0; j < 8; ++j) {
      const int row = j + half * 8;                   // timestep within chunk
      const float xsc = sigf(accg[j]) * tanhfast(accv[j]);
      const float av  = 0.001f + 0.998f * sigf(accd[j]);
      Xs[w][row][l16] = xsc;
      As[w][row][l16] = av;
    }
    __syncthreads();

    // --- causal scan over 16 timesteps; lanes 0..15 own one column --------
    if (lane < 16) {
      float h = hcarry;
#pragma unroll 4
      for (int j = 0; j < 16; ++j) {
        h = fmaf(As[w][j][lane], h, Xs[w][j][lane]);
        const int t = s0 + j;
        const float wt = gw[((size_t)b * SEQ + t) * NEXP + e];
        if (wt != 0.0f)
          atomicAdd(&out[((size_t)b * SEQ + t) * DMODEL + colbase + lane],
                    wt * h);
      }
      hcarry = h;
    }
  }
}

// ---------------------------------------------------------------------------
extern "C" void kernel_launch(void* const* d_in, const int* in_sizes, int n_in,
                              void* d_out, int out_size, void* d_ws, size_t ws_size,
                              hipStream_t stream) {
  (void)in_sizes; (void)n_in; (void)ws_size;
  const float* x     = (const float*)d_in[0];
  const float* Wg    = (const float*)d_in[1];
  const float* bg    = (const float*)d_in[2];
  const float* Wv    = (const float*)d_in[3];
  const float* bv    = (const float*)d_in[4];
  const float* Wd    = (const float*)d_in[5];
  const float* bd    = (const float*)d_in[6];
  const float* Wgate = (const float*)d_in[7];
  float* out = (float*)d_out;

  // Workspace layout (bytes):
  //   [0)            router weights gw : BATCH*SEQ*NEXP fp32   = 256 KB
  //   [0x40000)      x in bf16         : BATCH*SEQ*DMODEL bf16 = 32 MB
  //   [+32MB)        Wg,Wv,Wd in bf16  : 3 x NEXP*D*D bf16     = 24 MB
  char* ws = (char*)d_ws;
  float*    gw   = (float*)ws;
  uint16_t* xbf  = (uint16_t*)(ws + 0x40000);
  uint16_t* wgbf = (uint16_t*)(ws + 0x40000 + (size_t)BATCH * SEQ * DMODEL * 2);
  uint16_t* wvbf = wgbf + (size_t)NEXP * DMODEL * DMODEL;
  uint16_t* wdbf = wvbf + (size_t)NEXP * DMODEL * DMODEL;

  hipMemsetAsync(out, 0, (size_t)out_size * sizeof(float), stream);

  const int xw4 = BATCH * SEQ * DMODEL / 4;
  const int ww4 = NEXP * DMODEL * DMODEL / 4;
  cvt_bf16_kernel<<<(xw4 + 255) / 256, 256, 0, stream>>>(x,  xbf,  xw4);
  cvt_bf16_kernel<<<(ww4 + 255) / 256, 256, 0, stream>>>(Wg, wgbf, ww4);
  cvt_bf16_kernel<<<(ww4 + 255) / 256, 256, 0, stream>>>(Wv, wvbf, ww4);
  cvt_bf16_kernel<<<(ww4 + 255) / 256, 256, 0, stream>>>(Wd, wdbf, ww4);

  router_topk_kernel<<<(BATCH * SEQ) / 8, 256, 0, stream>>>(x, Wgate, gw);

  moe_mingru_expert_kernel<<<NEXP * BATCH * (DMODEL / 64), 128, 0, stream>>>(
      (const __bf16*)xbf, (const __bf16*)wgbf, (const __bf16*)wvbf,
      (const __bf16*)wdbf, bg, bv, bd, gw, out);
}
